// NystromTransformerLayer_27582279975354
// MI455X (gfx1250) — matrix operand order repaired and checked by reading, not compiled, not run on hardware
//
#include <hip/hip_runtime.h>
#include <hip/hip_bf16.h>

// ---------------------------------------------------------------------------
// Nystromformer layer for MI455X (gfx1250, wave32, WMMA).
// All GEMMs run on v_wmma_f32_16x16x32_f16 (f16 in, f32 accum).
// Shapes: B=4, N=4096, D=512, H=8, dh=64, M=256 landmarks, l=16.
// Every GEMM dim is a multiple of 64 (M,N) / 32 (K) -> no bounds checks.
// NN-path GEMMs stage their B tile through LDS (async global->LDS when the
// toolchain exposes the gfx1250 builtins, plain vector stores otherwise).
// ---------------------------------------------------------------------------

typedef _Float16 f16;
typedef __attribute__((ext_vector_type(8)))  _Float16 v8h;
typedef __attribute__((ext_vector_type(16))) _Float16 v16h;
typedef __attribute__((ext_vector_type(8)))  float    v8f;

#define DEV __device__ __forceinline__

#if defined(__gfx1250__) && __has_builtin(__builtin_amdgcn_global_load_async_to_lds_b128) && __has_builtin(__builtin_amdgcn_s_wait_asynccnt)
#define USE_ASYNC_LDS 1
// Probe-derived signature: (int4 addrspace(1)* src, int4 addrspace(3)* dst, imm, imm)
typedef int i4v __attribute__((vector_size(16)));
#define AS1_I4(p) ((__attribute__((address_space(1))) i4v*)(p))
#define AS3_I4(p) ((__attribute__((address_space(3))) i4v*)(p))
#else
#define USE_ASYNC_LDS 0
#endif

// ---- WMMA fragment helpers (rows/cols per CDNA5 ISA 7.12.2; contiguous k order in A and B, wave32) ----------
// A 16x32 f16: lane L holds row m0+L%16; elements e -> k = half*16 + e.
// Two contiguous 16B loads.
DEV v16h frag_a(const f16* A, int lda, int m0, int k0, int lane) {
  const int half = lane >> 4;
  const f16* p = A + (size_t)(m0 + (lane & 15)) * lda + k0 + half * 16;
  v8h lo = *(const v8h*)p;
  v8h hi = *(const v8h*)(p + 8);
  v16h r;
#pragma unroll
  for (int i = 0; i < 8; ++i) { r[i] = lo[i]; r[i + 8] = hi[i]; }
  return r;
}

// B 32x16 f16 when B stored transposed (N x K row-major, K contiguous):
// lane L holds col n=L%16, elements e -> k = e + 16*half. 16 contiguous f16.
DEV v16h frag_b_nt(const f16* Bt, int ldb, int n0, int k0, int lane) {
  const int half = lane >> 4;
  const f16* p = Bt + (size_t)(n0 + (lane & 15)) * ldb + k0 + half * 16;
  v8h lo = *(const v8h*)p;
  v8h hi = *(const v8h*)(p + 8);
  v16h r;
#pragma unroll
  for (int i = 0; i < 8; ++i) { r[i] = lo[i]; r[i + 8] = hi[i]; }
  return r;
}

// B fragment from an LDS tile holding a natural (K x N) 32x64 f16 block:
// strided per-lane gathers hit LDS instead of HBM/L2.
DEV v16h frag_b_lds(const f16* lb, int n0, int lane) {
  const int half = lane >> 4;
  const f16* p = lb + (half * 16) * 64 + n0 + (lane & 15);
  v16h r;
#pragma unroll
  for (int i = 0; i < 16; ++i) r[i] = p[i * 64];
  return r;
}

DEV v8f wmma16(v16h a, v16h b, v8f c) {
  return __builtin_amdgcn_wmma_f32_16x16x32_f16(
      /*neg_a=*/false, a, /*neg_b=*/false, b,
      /*c_mod=*/(short)0, c, /*reuse_a=*/false, /*reuse_b=*/false);
}

// ---------------------------------------------------------------------------
// Generic batched GEMM: C[M,N] = alpha * A[M,K] (*) B, block tile 64x64,
// 4 waves in 2x2 (each 32x32 = 4 wmma per K-step of 32).
//   BT=true : B stored transposed (N x K), fully vectorized global loads.
//   BT=false: B natural (K x N); 32x64 B tile staged into LDS per K-step
//             (async global->LDS when available), shared by all 4 waves.
// Epilogue writes any subset of:
//   Cplain16 = (f16) v              (v = alpha*acc (+ Cadd))
//   Cdiag16  = (f16)(diag*I - v)    (Newton-Schulz terms)
//   C32      = v
// ---------------------------------------------------------------------------
template <bool BT>
__global__ void __launch_bounds__(128)
gemm_kernel(const f16* __restrict__ A, const f16* __restrict__ B,
            f16* __restrict__ Cplain, f16* __restrict__ Cdiag,
            float* __restrict__ C32, const float* __restrict__ Cadd,
            int M, int N, int K, int lda, int ldb, int ldc,
            long long sA, long long sB, long long sC, long long sAdd,
            float alpha, float diagval) {
  (void)M; (void)N;
  const int batch = blockIdx.z;
  A += (size_t)batch * (size_t)sA;
  B += (size_t)batch * (size_t)sB;
  const int lane = threadIdx.x & 31;
  const int wave = threadIdx.x >> 5;
  const int wr = wave >> 1, wc = wave & 1;
  const int m0 = blockIdx.y * 64 + wr * 32;
  const int n0 = blockIdx.x * 64 + wc * 32;

  v8f zero = {0.f, 0.f, 0.f, 0.f, 0.f, 0.f, 0.f, 0.f};
  v8f acc[2][2] = {{zero, zero}, {zero, zero}};

  if constexpr (BT) {
    for (int k0 = 0; k0 < K; k0 += 32) {
      if (k0 + 32 < K)  // gfx1250 global_prefetch of the next K chunk
        __builtin_prefetch(A + (size_t)(m0 + (lane & 15)) * lda + k0 + 32, 0, 3);
      v16h a0 = frag_a(A, lda, m0, k0, lane);
      v16h a1 = frag_a(A, lda, m0 + 16, k0, lane);
      v16h b0 = frag_b_nt(B, ldb, n0, k0, lane);
      v16h b1 = frag_b_nt(B, ldb, n0 + 16, k0, lane);
      acc[0][0] = wmma16(a0, b0, acc[0][0]);
      acc[0][1] = wmma16(a0, b1, acc[0][1]);
      acc[1][0] = wmma16(a1, b0, acc[1][0]);
      acc[1][1] = wmma16(a1, b1, acc[1][1]);
    }
  } else {
    __shared__ __align__(16) f16 lb[32 * 64];  // 4 KiB B tile (K x N block)
    const int nblk = blockIdx.x * 64;
    const int t = threadIdx.x;
    const int kk = t >> 2;             // 0..31
    const int cc = (t & 3) << 4;       // 0,16,32,48
    const int nl0 = wc * 32;           // LDS-relative col of this wave's tile
    for (int k0 = 0; k0 < K; k0 += 32) {
      // ---- stage 32x64 B tile: 128 threads x 32B, coalesced ----
      const f16* gp = B + (size_t)(k0 + kk) * ldb + nblk + cc;
      f16* lp = &lb[kk * 64 + cc];
#if USE_ASYNC_LDS
      __builtin_amdgcn_global_load_async_to_lds_b128(AS1_I4(gp), AS3_I4(lp), 0, 0);
      __builtin_amdgcn_global_load_async_to_lds_b128(AS1_I4(gp + 8), AS3_I4(lp + 8), 0, 0);
      __builtin_amdgcn_s_wait_asynccnt(0);
#else
      *(v8h*)lp       = *(const v8h*)gp;
      *((v8h*)lp + 1) = *((const v8h*)gp + 1);
#endif
      __syncthreads();
      if (k0 + 32 < K)
        __builtin_prefetch(A + (size_t)(m0 + (lane & 15)) * lda + k0 + 32, 0, 3);
      v16h a0 = frag_a(A, lda, m0, k0, lane);
      v16h a1 = frag_a(A, lda, m0 + 16, k0, lane);
      v16h b0 = frag_b_lds(lb, nl0, lane);
      v16h b1 = frag_b_lds(lb, nl0 + 16, lane);
      acc[0][0] = wmma16(a0, b0, acc[0][0]);
      acc[0][1] = wmma16(a0, b1, acc[0][1]);
      acc[1][0] = wmma16(a1, b0, acc[1][0]);
      acc[1][1] = wmma16(a1, b1, acc[1][1]);
      __syncthreads();  // before next iteration overwrites the tile
    }
  }

  // C/D layout: lane -> col n0+.. + lane%16; VGPR r -> row +r+8*(lane/16).
  const int half = lane >> 4, ln = lane & 15;
#pragma unroll
  for (int i = 0; i < 2; ++i)
#pragma unroll
    for (int j = 0; j < 2; ++j)
#pragma unroll
      for (int r = 0; r < 8; ++r) {
        const int row = m0 + i * 16 + half * 8 + r;
        const int col = n0 + j * 16 + ln;
        const size_t idx = (size_t)batch * (size_t)sC + (size_t)row * ldc + col;
        float v = alpha * acc[i][j][r];
        if (Cadd) v += Cadd[(size_t)batch * (size_t)sAdd + (size_t)row * ldc + col];
        if (C32)    C32[idx] = v;
        if (Cplain) Cplain[idx] = (f16)v;
        if (Cdiag)  Cdiag[idx] = (f16)((row == col ? diagval : 0.0f) - v);
      }
}

// ---------------------------------------------------------------------------
// Output projection: Y = attn @ Wout + b_out + X  (M=16384, N=512, K=512).
// attn stored (b,h,n,64) f16; K index = h*64 + c. 8-wide chunks at 8-aligned
// offsets never cross a head boundary, so per-chunk head is uniform.
// ---------------------------------------------------------------------------
DEV v16h frag_a_attn(const f16* attn, int m0, int k0, int lane) {
  const int ln = lane & 15, half = lane >> 4;
  const int row = m0 + ln;
  const int b = row >> 12, n = row & 4095;
  const int h = k0 >> 6;
  const int cb = (k0 & 63) + half * 16;
  const f16* p = attn + (((size_t)(b * 8 + h) * 4096 + n) * 64) + cb;
  v8h lo = *(const v8h*)p;
  v8h hi = *(const v8h*)(p + 8);
  v16h r;
#pragma unroll
  for (int i = 0; i < 8; ++i) { r[i] = lo[i]; r[i + 8] = hi[i]; }
  return r;
}

__global__ void __launch_bounds__(128)
final_gemm_kernel(const f16* __restrict__ attn, const f16* __restrict__ WoutT,
                  const float* __restrict__ bias, const float* __restrict__ X,
                  float* __restrict__ Y) {
  const int lane = threadIdx.x & 31;
  const int wave = threadIdx.x >> 5;
  const int wr = wave >> 1, wc = wave & 1;
  const int m0 = blockIdx.y * 64 + wr * 32;
  const int n0 = blockIdx.x * 64 + wc * 32;
  v8f zero = {0.f, 0.f, 0.f, 0.f, 0.f, 0.f, 0.f, 0.f};
  v8f acc[2][2] = {{zero, zero}, {zero, zero}};
#pragma unroll
  for (int k0 = 0; k0 < 512; k0 += 32) {
    v16h a0 = frag_a_attn(attn, m0, k0, lane);
    v16h a1 = frag_a_attn(attn, m0 + 16, k0, lane);
    v16h b0 = frag_b_nt(WoutT, 512, n0, k0, lane);
    v16h b1 = frag_b_nt(WoutT, 512, n0 + 16, k0, lane);
    acc[0][0] = wmma16(a0, b0, acc[0][0]);
    acc[0][1] = wmma16(a0, b1, acc[0][1]);
    acc[1][0] = wmma16(a1, b0, acc[1][0]);
    acc[1][1] = wmma16(a1, b1, acc[1][1]);
  }
  const int half = lane >> 4, ln = lane & 15;
#pragma unroll
  for (int i = 0; i < 2; ++i)
#pragma unroll
    for (int j = 0; j < 2; ++j)
#pragma unroll
      for (int r = 0; r < 8; ++r) {
        const int row = m0 + i * 16 + half * 8 + r;
        const int col = n0 + j * 16 + ln;
        const size_t idx = (size_t)row * 512 + col;
        Y[idx] = acc[i][j][r] + bias[col] + X[idx];
      }
}

// ---------------------------------------------------------------------------
// Elementwise / reduction kernels
// ---------------------------------------------------------------------------
__global__ void ln_kernel(const float* __restrict__ X, const float* __restrict__ g,
                          const float* __restrict__ bt, f16* __restrict__ Xn) {
  const size_t row = blockIdx.x;  // 16384 rows of 512
  const float* x = X + row * 512;
  __shared__ float red[256];
  const int t = threadIdx.x;
  const float a0 = x[t], a1 = x[t + 256];
  red[t] = a0 + a1;
  __syncthreads();
  for (int o = 128; o > 0; o >>= 1) { if (t < o) red[t] += red[t + o]; __syncthreads(); }
  const float mu = red[0] * (1.0f / 512.0f);
  __syncthreads();
  const float d0 = a0 - mu, d1 = a1 - mu;
  red[t] = d0 * d0 + d1 * d1;
  __syncthreads();
  for (int o = 128; o > 0; o >>= 1) { if (t < o) red[t] += red[t + o]; __syncthreads(); }
  const float rs = rsqrtf(red[0] * (1.0f / 512.0f) + 1e-5f);
  Xn[row * 512 + t]       = (f16)(d0 * rs * g[t] + bt[t]);
  Xn[row * 512 + t + 256] = (f16)(d1 * rs * g[t + 256] + bt[t + 256]);
}

// dst[c][r] = (f16) src[r][c]   (rows x cols -> cols x rows)
__global__ void transpose_f32_to_f16(const float* __restrict__ src, f16* __restrict__ dst,
                                     int rows, int cols) {
  const int c = blockIdx.x * 16 + threadIdx.x;
  const int r = blockIdx.y * 16 + threadIdx.y;
  if (r < rows && c < cols) dst[(size_t)c * rows + r] = (f16)src[(size_t)r * cols + c];
}

// qkv (b,n,1536) -> q/k/v (b,h,n,64); q scaled by dh^-0.5 = 1/8
__global__ void scatter_qkv(const f16* __restrict__ qkv, f16* __restrict__ q,
                            f16* __restrict__ k, f16* __restrict__ v) {
  const size_t idx = (size_t)blockIdx.x * 256 + threadIdx.x;  // 4*4096*1536
  const int j = (int)(idx % 1536);
  const size_t bn = idx / 1536;
  const int n = (int)(bn & 4095);
  const int b = (int)(bn >> 12);
  const int which = j >> 9;  // /512
  const int d = j & 511;
  const int h = d >> 6, c = d & 63;
  const f16 val = qkv[idx];
  const size_t o = (((size_t)(b * 8 + h) * 4096) + n) * 64 + c;
  if (which == 0)      q[o] = (f16)((float)val * 0.125f);
  else if (which == 1) k[o] = val;
  else                 v[o] = val;
}

// landmark means over l=16 tokens
__global__ void landmark_kernel(const f16* __restrict__ q, const f16* __restrict__ k,
                                f16* __restrict__ ql, f16* __restrict__ kl) {
  const int c = threadIdx.x;            // 64
  const int bm = blockIdx.x;            // 32*256
  const int mi = bm & 255, bh = bm >> 8;
  const f16* qp = q + ((size_t)bh * 4096 + mi * 16) * 64 + c;
  const f16* kp = k + ((size_t)bh * 4096 + mi * 16) * 64 + c;
  float sq = 0.f, sk = 0.f;
#pragma unroll
  for (int t = 0; t < 16; ++t) { sq += (float)qp[t * 64]; sk += (float)kp[t * 64]; }
  const size_t o = ((size_t)bh * 256 + mi) * 64 + c;
  ql[o] = (f16)(sq * 0.0625f);
  kl[o] = (f16)(sk * 0.0625f);
}

// row-wise softmax: one 256-thread block per row of length L (256 or 4096)
__global__ void softmax_kernel(const float* __restrict__ src, f16* __restrict__ dst16,
                               float* __restrict__ dst32, int L) {
  const size_t row = blockIdx.x;
  const float* s = src + row * (size_t)L;
  __shared__ float red[256];
  const int t = threadIdx.x;
  float mx = -3.0e38f;
  for (int i = t; i < L; i += 256) mx = fmaxf(mx, s[i]);
  red[t] = mx; __syncthreads();
  for (int o = 128; o > 0; o >>= 1) { if (t < o) red[t] = fmaxf(red[t], red[t + o]); __syncthreads(); }
  mx = red[0]; __syncthreads();
  float sum = 0.f;
  for (int i = t; i < L; i += 256) sum += __expf(s[i] - mx);
  red[t] = sum; __syncthreads();
  for (int o = 128; o > 0; o >>= 1) { if (t < o) red[t] += red[t + o]; __syncthreads(); }
  const float inv = 1.0f / red[0];
  for (int i = t; i < L; i += 256) {
    const float v = __expf(s[i] - mx) * inv;
    dst16[row * (size_t)L + i] = (f16)v;
    if (dst32) dst32[row * (size_t)L + i] = v;
  }
}

// per-(b,h) max row/col L1 norms of a2 (for pinv init scale)
__global__ void pinv_norms(const float* __restrict__ a2, float* __restrict__ mc,
                           float* __restrict__ mr) {
  const int bh = blockIdx.x, t = threadIdx.x;  // 32 x 256
  const float* A = a2 + (size_t)bh * 65536;
  float cs = 0.f, rs = 0.f;
  for (int j = 0; j < 256; ++j) { cs += fabsf(A[t * 256 + j]); rs += fabsf(A[j * 256 + t]); }
  __shared__ float r1[256], r2[256];
  r1[t] = cs; r2[t] = rs; __syncthreads();
  for (int o = 128; o > 0; o >>= 1) {
    if (t < o) { r1[t] = fmaxf(r1[t], r1[t + o]); r2[t] = fmaxf(r2[t], r2[t + o]); }
    __syncthreads();
  }
  if (t == 0) { mc[bh] = r1[0]; mr[bh] = r2[0]; }
}

__global__ void pinv_scale(const float* __restrict__ mc, const float* __restrict__ mr,
                           float* __restrict__ slot) {
  if (threadIdx.x == 0) {
    float a = mc[0], b = mr[0];
    for (int i = 1; i < 32; ++i) { a = fmaxf(a, mc[i]); b = fmaxf(b, mr[i]); }
    slot[0] = 1.0f / (a * b);
  }
}

// z0 = a2^T * scale  (f16)
__global__ void z_init(const float* __restrict__ a2, const float* __restrict__ slot,
                       f16* __restrict__ z) {
  const size_t idx = (size_t)blockIdx.x * 256 + threadIdx.x;  // 32*256*256
  const int j = (int)(idx & 255);
  const int i = (int)((idx >> 8) & 255);
  const size_t bh = idx >> 16;
  z[idx] = (f16)(a2[(bh << 16) + ((size_t)j << 8) + i] * slot[0]);
}

// depthwise residual conv along the sequence dim, kernel 33, pad 16
__global__ void conv_kernel(const f16* __restrict__ v, const float* __restrict__ w,
                            float* __restrict__ res) {
  const size_t idx = (size_t)blockIdx.x * 256 + threadIdx.x;  // 4*8*4096*64
  const int c = (int)(idx & 63);
  const size_t r1 = idx >> 6;
  const int n = (int)(r1 & 4095);
  const size_t bh = r1 >> 12;
  const int h = (int)(bh & 7);
  const f16* vb = v + (bh << 18);  // 4096*64
  float acc = 0.f;
#pragma unroll
  for (int t = 0; t < 33; ++t) {
    const int nn = n + t - 16;
    if (nn >= 0 && nn < 4096) acc += w[h * 33 + t] * (float)vb[((size_t)nn << 6) + c];
  }
  res[idx] = acc;
}

// ---------------------------------------------------------------------------
// Host launch. Workspace arena (~353 MiB) with documented reuse.
// ---------------------------------------------------------------------------
extern "C" void kernel_launch(void* const* d_in, const int* in_sizes, int n_in,
                              void* d_out, int out_size, void* d_ws, size_t ws_size,
                              hipStream_t stream) {
  (void)in_sizes; (void)n_in; (void)out_size; (void)ws_size;
  const float* X     = (const float*)d_in[0];
  const float* Wqkv  = (const float*)d_in[1];
  const float* Wout  = (const float*)d_in[2];
  const float* bout  = (const float*)d_in[3];
  const float* convw = (const float*)d_in[4];
  const float* g1    = (const float*)d_in[5];
  const float* bt1   = (const float*)d_in[6];
  float* Y = (float*)d_out;
  char* ws = (char*)d_ws;

  // ---- arena ----
  const size_t oXn  = 0;                    // 16 MiB  (dead after QKV gemm)
  const size_t oQKV = oXn + 16777216;       // 48 MiB  (dead after scatter)
  const size_t oA1  = 0;                    // 64 MiB  a1 reuses Xn+QKV region
  size_t off = 67108864;
  const size_t oWqkvT = off; off += 1572864;   // Wqkv^T f16 (1536x512)
  const size_t oWoutT = off; off += 524288;    // Wout^T f16 (512x512)
  const size_t oQl    = off; off += 1048576;   // q landmarks (b,h,256,64) f16
  const size_t oKl    = off; off += 1048576;
  const size_t oQ     = off; off += 16777216;  // q (b,h,4096,64) f16 (scaled)
  const size_t oK     = off; off += 16777216;
  const size_t oV     = off; off += 16777216;
  const size_t oSim2  = off; off += 8388608;   // sim2 f32
  const size_t oA2f   = off; off += 8388608;   // a2 f32
  const size_t oA2h   = off; off += 4194304;   // a2 f16
  const size_t oZa    = off; off += 4194304;   // pinv ping-pong
  const size_t oZb    = off; off += 4194304;
  const size_t oXz    = off; off += 4194304;   // xz f16
  const size_t oU1    = off; off += 4194304;   // 7I - xz
  const size_t oU2    = off; off += 4194304;   // 15I - xz*u1
  const size_t oU3    = off; off += 4194304;   // 13I - xz*u2
  const size_t oKv    = off; off += 1048576;   // a3@v (b,h,256,64) f16
  const size_t oMc    = off; off += 256;
  const size_t oMr    = off; off += 256;
  const size_t oSlot  = off; off += 256;
  const size_t oBig   = off; off += 134217728; // sim1 then sim3 (f32)
  const size_t oA1z   = oBig;                  // 64 MiB, after sim3 consumed
  const size_t oRes   = oBig + 67108864;       // conv residual f32 (32 MiB)
  const size_t oAttn  = oBig + 100663296;      // attn f16 (b,h,4096,64), 16 MiB
  const size_t oA3    = off; off += 67108864;  // a3 f16

  f16*   Xn    = (f16*)(ws + oXn);
  f16*   qkv16 = (f16*)(ws + oQKV);
  f16*   a1h   = (f16*)(ws + oA1);
  f16*   WqkvT = (f16*)(ws + oWqkvT);
  f16*   WoutT = (f16*)(ws + oWoutT);
  f16*   ql    = (f16*)(ws + oQl);
  f16*   kl    = (f16*)(ws + oKl);
  f16*   q16   = (f16*)(ws + oQ);
  f16*   k16   = (f16*)(ws + oK);
  f16*   v16   = (f16*)(ws + oV);
  float* sim2  = (float*)(ws + oSim2);
  float* a2f   = (float*)(ws + oA2f);
  f16*   a2h   = (f16*)(ws + oA2h);
  f16*   zA    = (f16*)(ws + oZa);
  f16*   zB    = (f16*)(ws + oZb);
  f16*   xz    = (f16*)(ws + oXz);
  f16*   u1    = (f16*)(ws + oU1);
  f16*   u2    = (f16*)(ws + oU2);
  f16*   u3    = (f16*)(ws + oU3);
  f16*   kv16  = (f16*)(ws + oKv);
  float* mc    = (float*)(ws + oMc);
  float* mr    = (float*)(ws + oMr);
  float* slot  = (float*)(ws + oSlot);
  float* big   = (float*)(ws + oBig);
  f16*   a1z   = (f16*)(ws + oA1z);
  float* res32 = (float*)(ws + oRes);
  f16*   attn  = (f16*)(ws + oAttn);
  f16*   a3h   = (f16*)(ws + oA3);

  const dim3 blk(128);

  // 1. LayerNorm -> f16
  ln_kernel<<<16384, 256, 0, stream>>>(X, g1, bt1, Xn);
  // 2. weight transposes -> f16 (NT GEMM operands)
  transpose_f32_to_f16<<<dim3(96, 32), dim3(16, 16), 0, stream>>>(Wqkv, WqkvT, 512, 1536);
  transpose_f32_to_f16<<<dim3(32, 32), dim3(16, 16), 0, stream>>>(Wout, WoutT, 512, 512);
  // 3. QKV projection: (16384x512)x(512x1536)
  gemm_kernel<true><<<dim3(24, 256, 1), blk, 0, stream>>>(
      Xn, WqkvT, qkv16, nullptr, nullptr, nullptr,
      16384, 1536, 512, 512, 512, 1536, 0, 0, 0, 0, 1.0f, 0.0f);
  // 4. split to per-head q/k/v (q scaled)
  scatter_qkv<<<98304, 256, 0, stream>>>(qkv16, q16, k16, v16);
  // 5. landmarks
  landmark_kernel<<<8192, 64, 0, stream>>>(q16, k16, ql, kl);
  // 6. sim2 = q_l k_l^T (NT), then softmax -> a2 (f32+f16)
  gemm_kernel<true><<<dim3(4, 4, 32), blk, 0, stream>>>(
      ql, kl, nullptr, nullptr, sim2, nullptr,
      256, 256, 64, 64, 64, 256, 16384, 16384, 65536, 0, 1.0f, 0.0f);
  softmax_kernel<<<8192, 256, 0, stream>>>(sim2, a2h, a2f, 256);
  // 7. Moore-Penrose pinv (6 Newton-Schulz iterations), all batched WMMA GEMMs
  pinv_norms<<<32, 256, 0, stream>>>(a2f, mc, mr);
  pinv_scale<<<1, 32, 0, stream>>>(mc, mr, slot);
  z_init<<<8192, 256, 0, stream>>>(a2f, slot, zA);
  f16* zc = zA; f16* zn = zB;
  for (int it = 0; it < 6; ++it) {
    gemm_kernel<false><<<dim3(4, 4, 32), blk, 0, stream>>>(   // xz, u1 = 7I - xz
        a2h, zc, xz, u1, nullptr, nullptr,
        256, 256, 256, 256, 256, 256, 65536, 65536, 65536, 0, 1.0f, 7.0f);
    gemm_kernel<false><<<dim3(4, 4, 32), blk, 0, stream>>>(   // u2 = 15I - xz*u1
        xz, u1, nullptr, u2, nullptr, nullptr,
        256, 256, 256, 256, 256, 256, 65536, 65536, 65536, 0, 1.0f, 15.0f);
    gemm_kernel<false><<<dim3(4, 4, 32), blk, 0, stream>>>(   // u3 = 13I - xz*u2
        xz, u2, nullptr, u3, nullptr, nullptr,
        256, 256, 256, 256, 256, 256, 65536, 65536, 65536, 0, 1.0f, 13.0f);
    gemm_kernel<false><<<dim3(4, 4, 32), blk, 0, stream>>>(   // z' = 0.25 z*u3
        zc, u3, zn, nullptr, nullptr, nullptr,
        256, 256, 256, 256, 256, 256, 65536, 65536, 65536, 0, 0.25f, 0.0f);
    f16* t = zc; zc = zn; zn = t;
  }
  // 8. sim1 = q k_l^T (NT) -> softmax -> a1 (a1 reuses Xn+qkv region)
  gemm_kernel<true><<<dim3(4, 64, 32), blk, 0, stream>>>(
      q16, kl, nullptr, nullptr, big, nullptr,
      4096, 256, 64, 64, 64, 256, 262144, 16384, 1048576, 0, 1.0f, 0.0f);
  softmax_kernel<<<131072, 256, 0, stream>>>(big, a1h, nullptr, 256);
  // 9. sim3 = q_l k^T (NT) -> softmax -> a3
  gemm_kernel<true><<<dim3(64, 4, 32), blk, 0, stream>>>(
      ql, k16, nullptr, nullptr, big, nullptr,
      256, 4096, 64, 64, 64, 4096, 16384, 262144, 1048576, 0, 1.0f, 0.0f);
  softmax_kernel<<<8192, 256, 0, stream>>>(big, a3h, nullptr, 4096);
  // 10. kv = a3 @ v  (NN: B is (4096x64) natural, LDS-staged)
  gemm_kernel<false><<<dim3(1, 4, 32), blk, 0, stream>>>(
      a3h, v16, kv16, nullptr, nullptr, nullptr,
      256, 64, 4096, 4096, 64, 64, 1048576, 262144, 16384, 0, 1.0f, 0.0f);
  // 11. a1z = a1 @ pinv(a2)   (big region now free -> holds a1z)
  gemm_kernel<false><<<dim3(4, 64, 32), blk, 0, stream>>>(
      a1h, zc, a1z, nullptr, nullptr, nullptr,
      4096, 256, 256, 256, 256, 256, 1048576, 65536, 1048576, 0, 1.0f, 0.0f);
  // 12. depthwise conv residual on v
  conv_kernel<<<32768, 256, 0, stream>>>(v16, convw, res32);
  // 13. attn = a1z @ kv + conv_res   (b,h,4096,64) f16
  gemm_kernel<false><<<dim3(1, 64, 32), blk, 0, stream>>>(
      a1z, kv16, attn, nullptr, nullptr, res32,
      4096, 64, 256, 256, 64, 64, 1048576, 16384, 262144, 262144, 1.0f, 0.0f);
  // 14. Y = attn @ Wout + b_out + X
  final_gemm_kernel<<<dim3(8, 256, 1), blk, 0, stream>>>(attn, WoutT, bout, X, Y);
}
